// DGCNN_22462678958317
// MI455X (gfx1250) — matrix operand order repaired and checked by reading, not compiled, not run on hardware
//
#include <hip/hip_runtime.h>

typedef _Float16 f16;
typedef __attribute__((ext_vector_type(8)))  _Float16 v8h;
typedef __attribute__((ext_vector_type(16))) _Float16 v16h;
typedef __attribute__((ext_vector_type(8)))  float    v8f;

#define EPSV   1e-5f
#define SLOPE  0.2f
#define NPTS   4096
#define BATCH  2
#define KNB    20            // k neighbors
#define MTOT   (BATCH*NPTS)  // 8192 rows, batch folded into GEMM M
#define LDH    512
#define QB     128           // kNN queries per block
#define TK     32            // kNN key tile

// ---------------------------------------------------------------------------
// prep kernels
// ---------------------------------------------------------------------------
__global__ void k_transpose(const float* __restrict__ x,
                            float* __restrict__ xt32, f16* __restrict__ xt16) {
  int tid = blockIdx.x * blockDim.x + threadIdx.x;           // MTOT*32
  if (tid >= MTOT * 32) return;
  int c = tid & 31, p = tid >> 5;
  int b = p / NPTS, n = p - b * NPTS;
  float v = (c < 15) ? x[(b * 15 + c) * NPTS + n] : 0.f;
  xt32[tid] = v;
  xt16[tid] = (f16)v;
}

// W (O x 2C) -> W' (2O x Kp) f16: rows [0,O)=Wn, rows [O,2O)=Wc-Wn
__global__ void k_edge_w(const float* __restrict__ W, int O, int C, int Kp,
                         f16* __restrict__ outw) {
  int tid = blockIdx.x * blockDim.x + threadIdx.x;
  if (tid >= 2 * O * Kp) return;
  int row = tid / Kp, c = tid - row * Kp;
  float v = 0.f;
  if (c < C) {
    int o = (row < O) ? row : row - O;
    const float* wr = W + (size_t)o * 2 * C;
    v = (row < O) ? wr[c] : (wr[C + c] - wr[c]);
  }
  outw[tid] = (f16)v;
}

// W (O x C) -> f16 padded (Op x Kp)
__global__ void k_point_w(const float* __restrict__ W, int O, int C, int Op,
                          int Kp, f16* __restrict__ outw) {
  int tid = blockIdx.x * blockDim.x + threadIdx.x;
  if (tid >= Op * Kp) return;
  int r = tid / Kp, c = tid - r * Kp;
  outw[tid] = (f16)((r < O && c < C) ? W[(size_t)r * C + c] : 0.f);
}

// fold BN: s = g*rsqrt(v+eps); bias = b - m*s
__global__ void k_bn(const float* __restrict__ g, const float* __restrict__ b,
                     const float* __restrict__ m, const float* __restrict__ v,
                     int O, float* __restrict__ sc, float* __restrict__ bi) {
  int tid = blockIdx.x * blockDim.x + threadIdx.x;
  if (tid >= O) return;
  float s = g[tid] * rsqrtf(v[tid] + EPSV);
  sc[tid] = s;
  bi[tid] = b[tid] - m[tid] * s;
}

__global__ void k_sqnorm(const float* __restrict__ F, int ld, int C,
                         float* __restrict__ xx) {
  int p = blockIdx.x * blockDim.x + threadIdx.x;
  if (p >= MTOT) return;
  const float* r = F + (size_t)p * ld;
  float s = 0.f;
  for (int c = 0; c < C; c += 4) {
    float4 q = *(const float4*)(r + c);
    s += q.x * q.x + q.y * q.y + q.z * q.z + q.w * q.w;
  }
  xx[p] = s;
}

// ---------------------------------------------------------------------------
// fused pairwise-distance + top-k (largest pd = nearest), matches lax.top_k
// tie behavior (lower index first) via strict '>' with ascending key scan.
// Key tiles staged global->LDS with GLOBAL_LOAD_ASYNC_TO_LDS_B128 (ASYNCcnt).
// ---------------------------------------------------------------------------
__global__ void k_knn(const float* __restrict__ F, int ldf, int C,
                      const float* __restrict__ xx, int* __restrict__ idx) {
  __shared__ float kt[TK * 256];
  __shared__ float xs[TK];
  int b = blockIdx.y;
  int q = blockIdx.x * QB + threadIdx.x;
  int p = b * NPTS + q;
  float qq = xx[p];
  float best[KNB];
  int   bidx[KNB];
#pragma unroll
  for (int i = 0; i < KNB; i++) { best[i] = -3.0e38f; bidx[i] = 0; }
  const float* qrow = F + (size_t)p * ldf;
  int c4n = C >> 2;
  for (int j0 = 0; j0 < NPTS; j0 += TK) {
    __syncthreads();   // protect kt from previous iteration's readers
    int elems = TK * c4n;  // number of 16B chunks in the tile
    for (int e = threadIdx.x; e < elems; e += QB) {
      int j = e / c4n, c4 = e - j * c4n;
      const float* gsrc = F + (size_t)(b * NPTS + j0 + j) * ldf + c4 * 4;
      unsigned lds_off = (unsigned)(size_t)(kt + j * C + c4 * 4);
      // CDNA5 async copy: global -> LDS, no VGPR round-trip, ASYNCcnt-tracked
      asm volatile("global_load_async_to_lds_b128 %0, %1, off"
                   :: "v"(lds_off), "v"(gsrc) : "memory");
    }
    if (threadIdx.x < TK) xs[threadIdx.x] = xx[b * NPTS + j0 + threadIdx.x];
    asm volatile("s_wait_asynccnt 0" ::: "memory");
    __syncthreads();
    float acc[TK];
#pragma unroll
    for (int j = 0; j < TK; j++) acc[j] = 0.f;
    for (int c4 = 0; c4 < c4n; c4++) {
      float4 qv = *(const float4*)(qrow + c4 * 4);
#pragma unroll
      for (int j = 0; j < TK; j++) {
        float4 kv = *(const float4*)(kt + j * C + c4 * 4);
        acc[j] += qv.x * kv.x + qv.y * kv.y + qv.z * kv.z + qv.w * kv.w;
      }
    }
#pragma unroll
    for (int j = 0; j < TK; j++) {
      float d = 2.f * acc[j] - qq - xs[j];
      int jj = j0 + j;
      if (d > best[KNB - 1]) {
        // predicated shift network: all indices constant -> stays in VGPRs
#pragma unroll
        for (int s = KNB - 1; s >= 1; s--) {
          bool gt  = d > best[s - 1];
          float nb = gt ? best[s - 1] : d;
          int   ni = gt ? bidx[s - 1] : jj;
          if (d > best[s]) { best[s] = nb; bidx[s] = ni; }
        }
        if (d > best[0]) { best[0] = d; bidx[0] = jj; }
      }
    }
  }
#pragma unroll
  for (int i = 0; i < KNB; i++) idx[(size_t)p * KNB + i] = bidx[i];
}

// ---------------------------------------------------------------------------
// WMMA GEMM: Y[M x Ncols] = A[M x K] (f16, lda) * Bw[Ncols x K]^T (f16, ldb)
// One wave computes a (16*MT) x (16*NT) tile. Per K-step all fragments are
// loaded first (one clause/wait), then MT*NT independent
// v_wmma_f32_16x16x32_f16 run back-to-back (no D->A/B hazards).
// MT=2,NT=2: 8 b128 loads per 4 WMMAs, ~90 live VGPRs -> no scratch spills.
// Optional fused BN + leaky relu epilogue.
// ---------------------------------------------------------------------------
template <int MT, int NT>
__global__ void k_gemm(const f16* __restrict__ A, int lda,
                       const f16* __restrict__ Bw, int ldb, int K,
                       float* __restrict__ yf, f16* __restrict__ yh, int ldy,
                       const float* __restrict__ scale,
                       const float* __restrict__ bias, int do_lrelu) {
  int lane = threadIdx.x;           // 0..31 (wave32)
  int hi = lane >> 4, l = lane & 15;
  int mrow0 = blockIdx.x * 16 * MT;
  int ncol0 = blockIdx.y * 16 * NT;
  const f16* arow[MT];
  const f16* brow[NT];
  v8f acc[MT][NT];
#pragma unroll
  for (int m = 0; m < MT; m++) {
    arow[m] = A + (size_t)(mrow0 + m * 16 + l) * lda + hi * 8;
#pragma unroll
    for (int t = 0; t < NT; t++)
      acc[m][t] = (v8f){0.f, 0.f, 0.f, 0.f, 0.f, 0.f, 0.f, 0.f};
  }
#pragma unroll
  for (int t = 0; t < NT; t++)
    brow[t] = Bw + (size_t)(ncol0 + t * 16 + l) * ldb + hi * 8;

  for (int k0 = 0; k0 < K; k0 += 32) {
    v16h a[MT], bm[NT];
    // A 16x32 frag: lanes 0-15 -> K {0..7,16..23}; lanes 16-31 -> K {8..15,24..31}
#pragma unroll
    for (int m = 0; m < MT; m++) {
      v8h a0 = *(const v8h*)(arow[m] + k0);
      v8h a1 = *(const v8h*)(arow[m] + k0 + 16);
      a[m] = __builtin_shufflevector(a0, a1, 0,1,2,3,4,5,6,7,8,9,10,11,12,13,14,15);
    }
    // B 32x16 frag: lanes 0-15 -> K {0..7,16..23} of col l; lanes 16-31 -> K {8..15,24..31}
#pragma unroll
    for (int t = 0; t < NT; t++) {
      v8h b0 = *(const v8h*)(brow[t] + k0);
      v8h b1 = *(const v8h*)(brow[t] + k0 + 16);
      bm[t] = __builtin_shufflevector(b0, b1, 0,1,2,3,4,5,6,7,8,9,10,11,12,13,14,15);
    }
#pragma unroll
    for (int m = 0; m < MT; m++)
#pragma unroll
      for (int t = 0; t < NT; t++)
        acc[m][t] = __builtin_amdgcn_wmma_f32_16x16x32_f16(
            false, a[m], false, bm[t], (short)0, acc[m][t], false, false);
  }
#pragma unroll
  for (int t = 0; t < NT; t++) {
    int col = ncol0 + t * 16 + l;
    float s = 1.f, bi = 0.f;
    if (scale) { s = scale[col]; bi = bias[col]; }
#pragma unroll
    for (int m = 0; m < MT; m++) {
#pragma unroll
      for (int r = 0; r < 8; r++) {
        int row = mrow0 + m * 16 + r + hi * 8;  // D: VGPR r -> M=r / M=r+8
        float v = acc[m][t][r];
        if (scale) v = v * s + bi;
        if (do_lrelu) v = (v >= 0.f) ? v : SLOPE * v;
        size_t o = (size_t)row * ldy + col;
        if (yf) yf[o] = v;
        if (yh) yh[o] = (f16)v;
      }
    }
  }
}

// ---------------------------------------------------------------------------
// edge aggregate: out[n,o] = max_kk lrelu(bn(T[idx[kk], o] + T[n, O+o]))
// writes into concat feature buffer (f32 for next kNN, f16 for next GEMM A)
// ---------------------------------------------------------------------------
__global__ void k_agg(const float* __restrict__ T, int O,
                      const int* __restrict__ idx,
                      const float* __restrict__ sc, const float* __restrict__ bi,
                      float* __restrict__ Hf, f16* __restrict__ Hh, int coff) {
  int og4 = O >> 2;
  int tid = blockIdx.x * blockDim.x + threadIdx.x;
  if (tid >= MTOT * og4) return;
  int p = tid / og4, g = (tid - p * og4) << 2;
  int b = p / NPTS;
  int O2 = 2 * O;
  const float* t1p = T + (size_t)p * O2 + O + g;
  float t10 = t1p[0], t11 = t1p[1], t12 = t1p[2], t13 = t1p[3];
  float s0 = sc[g], s1 = sc[g+1], s2 = sc[g+2], s3 = sc[g+3];
  float c0 = bi[g], c1 = bi[g+1], c2 = bi[g+2], c3 = bi[g+3];
  float m0 = -3e38f, m1 = -3e38f, m2 = -3e38f, m3 = -3e38f;
  const int* ip = idx + (size_t)p * KNB;
  for (int kk = 0; kk < KNB; kk++) {
    int j = ip[kk];
    const float* t2 = T + (size_t)(b * NPTS + j) * O2 + g;
    float y0 = (t2[0] + t10) * s0 + c0; y0 = (y0 >= 0.f) ? y0 : SLOPE * y0; m0 = fmaxf(m0, y0);
    float y1 = (t2[1] + t11) * s1 + c1; y1 = (y1 >= 0.f) ? y1 : SLOPE * y1; m1 = fmaxf(m1, y1);
    float y2 = (t2[2] + t12) * s2 + c2; y2 = (y2 >= 0.f) ? y2 : SLOPE * y2; m2 = fmaxf(m2, y2);
    float y3 = (t2[3] + t13) * s3 + c3; y3 = (y3 >= 0.f) ? y3 : SLOPE * y3; m3 = fmaxf(m3, y3);
  }
  size_t o = (size_t)p * LDH + coff + g;
  Hf[o]   = m0; Hf[o+1] = m1; Hf[o+2] = m2; Hf[o+3] = m3;
  Hh[o]   = (f16)m0; Hh[o+1] = (f16)m1; Hh[o+2] = (f16)m2; Hh[o+3] = (f16)m3;
}

// (B,N,16) -> d_out (B,2,N), keep channels 0..1
__global__ void k_writeout(const float* __restrict__ Y8, float* __restrict__ out) {
  int tid = blockIdx.x * blockDim.x + threadIdx.x;   // BATCH*2*NPTS
  if (tid >= BATCH * 2 * NPTS) return;
  int b = tid / (2 * NPTS);
  int r = tid - b * 2 * NPTS;
  int o = r / NPTS, n = r - o * NPTS;
  out[tid] = Y8[(size_t)(b * NPTS + n) * 16 + o];
}

// ---------------------------------------------------------------------------
extern "C" void kernel_launch(void* const* d_in, const int* in_sizes, int n_in,
                              void* d_out, int out_size, void* d_ws, size_t ws_size,
                              hipStream_t stream) {
  (void)in_sizes; (void)n_in; (void)out_size; (void)ws_size;
  const float* x  = (const float*)d_in[0];
  const float* W1 = (const float*)d_in[1];
  const float* W2 = (const float*)d_in[2];
  const float* W3 = (const float*)d_in[3];
  const float* W4 = (const float*)d_in[4];
  const float* W5 = (const float*)d_in[5];
  const float* W6 = (const float*)d_in[6];
  const float* W7 = (const float*)d_in[7];
  const float* W8 = (const float*)d_in[8];
  const float *gp[7], *bp[7], *mp[7], *vp[7];
  for (int i = 0; i < 7; i++) {
    gp[i] = (const float*)d_in[9  + 2*i];
    bp[i] = (const float*)d_in[10 + 2*i];
    mp[i] = (const float*)d_in[23 + 2*i];
    vp[i] = (const float*)d_in[24 + 2*i];
  }
  float* out = (float*)d_out;

  char* ws = (char*)d_ws;
  size_t off = 0;
  auto alloc = [&](size_t bytes) -> char* {
    char* p = ws + off;
    off = (off + bytes + 255) & ~(size_t)255;
    return p;
  };
  float* Xt32 = (float*)alloc((size_t)MTOT * 32 * 4);
  f16*   Xt16 = (f16*)  alloc((size_t)MTOT * 32 * 2);
  float* Hf   = (float*)alloc((size_t)MTOT * 512 * 4);
  f16*   Hh   = (f16*)  alloc((size_t)MTOT * 512 * 2);
  float* T    = (float*)alloc((size_t)MTOT * 512 * 4);
  int*   idx  = (int*)  alloc((size_t)MTOT * KNB * 4);
  float* xx   = (float*)alloc((size_t)MTOT * 4);
  f16*   P5h  = (f16*)  alloc((size_t)MTOT * 256 * 2);
  f16*   P6h  = (f16*)  alloc((size_t)MTOT * 128 * 2);
  f16*   P7h  = (f16*)  alloc((size_t)MTOT * 64 * 2);
  float* Y8   = (float*)alloc((size_t)MTOT * 16 * 4);
  f16* We1 = (f16*)alloc(128 * 32 * 2);
  f16* We2 = (f16*)alloc(128 * 64 * 2);
  f16* We3 = (f16*)alloc(256 * 64 * 2);
  f16* We4 = (f16*)alloc(512 * 128 * 2);
  f16* Wp5 = (f16*)alloc(256 * 512 * 2);
  f16* Wp6 = (f16*)alloc(128 * 256 * 2);
  f16* Wp7 = (f16*)alloc(64 * 128 * 2);
  f16* Wp8 = (f16*)alloc(16 * 64 * 2);
  float* SC = (float*)alloc(7 * 256 * 4);
  float* BI = (float*)alloc(7 * 256 * 4);

  // ---- parameter prep ----
  const int bnO[7] = {64, 64, 128, 256, 256, 128, 64};
  for (int i = 0; i < 7; i++)
    k_bn<<<(bnO[i] + 63) / 64, 64, 0, stream>>>(gp[i], bp[i], mp[i], vp[i],
                                                bnO[i], SC + i * 256, BI + i * 256);
  k_edge_w<<<(2*64*32   + 255)/256, 256, 0, stream>>>(W1, 64,  15,  32,  We1);
  k_edge_w<<<(2*64*64   + 255)/256, 256, 0, stream>>>(W2, 64,  64,  64,  We2);
  k_edge_w<<<(2*128*64  + 255)/256, 256, 0, stream>>>(W3, 128, 64,  64,  We3);
  k_edge_w<<<(2*256*128 + 255)/256, 256, 0, stream>>>(W4, 256, 128, 128, We4);
  k_point_w<<<(256*512 + 255)/256, 256, 0, stream>>>(W5, 256, 512, 256, 512, Wp5);
  k_point_w<<<(128*256 + 255)/256, 256, 0, stream>>>(W6, 128, 256, 128, 256, Wp6);
  k_point_w<<<(64*128  + 255)/256, 256, 0, stream>>>(W7, 64,  128, 64,  128, Wp7);
  k_point_w<<<(16*64   + 255)/256, 256, 0, stream>>>(W8, 2,   64,  16,  64,  Wp8);
  k_transpose<<<(MTOT * 32 + 255) / 256, 256, 0, stream>>>(x, Xt32, Xt16);

  dim3 knng(NPTS / QB, BATCH);
  const int M32 = MTOT / 32;   // 256 blocks in M at MT=2

  // ---- edge block 1: feat Xt (C=16 padded, lda 32), O=64 ----
  k_sqnorm<<<(MTOT + 255) / 256, 256, 0, stream>>>(Xt32, 32, 16, xx);
  k_knn<<<knng, QB, 0, stream>>>(Xt32, 32, 16, xx, idx);
  k_gemm<2,2><<<dim3(M32, 4), 32, 0, stream>>>(Xt16, 32, We1, 32, 32,
                                               T, (f16*)nullptr, 128,
                                               nullptr, nullptr, 0);
  k_agg<<<(MTOT * 16 + 255) / 256, 256, 0, stream>>>(T, 64, idx, SC, BI, Hf, Hh, 0);

  // ---- edge block 2: feat H[:,0:64], O=64 ----
  k_sqnorm<<<(MTOT + 255) / 256, 256, 0, stream>>>(Hf, 512, 64, xx);
  k_knn<<<knng, QB, 0, stream>>>(Hf, 512, 64, xx, idx);
  k_gemm<2,2><<<dim3(M32, 4), 32, 0, stream>>>(Hh, 512, We2, 64, 64,
                                               T, (f16*)nullptr, 128,
                                               nullptr, nullptr, 0);
  k_agg<<<(MTOT * 16 + 255) / 256, 256, 0, stream>>>(T, 64, idx, SC + 256, BI + 256,
                                                     Hf, Hh, 64);

  // ---- edge block 3: feat H[:,64:128], O=128 ----
  k_sqnorm<<<(MTOT + 255) / 256, 256, 0, stream>>>(Hf + 64, 512, 64, xx);
  k_knn<<<knng, QB, 0, stream>>>(Hf + 64, 512, 64, xx, idx);
  k_gemm<2,2><<<dim3(M32, 8), 32, 0, stream>>>(Hh + 64, 512, We3, 64, 64,
                                               T, (f16*)nullptr, 256,
                                               nullptr, nullptr, 0);
  k_agg<<<(MTOT * 32 + 255) / 256, 256, 0, stream>>>(T, 128, idx, SC + 512, BI + 512,
                                                     Hf, Hh, 128);

  // ---- edge block 4: feat H[:,128:256], O=256 ----
  k_sqnorm<<<(MTOT + 255) / 256, 256, 0, stream>>>(Hf + 128, 512, 128, xx);
  k_knn<<<knng, QB, 0, stream>>>(Hf + 128, 512, 128, xx, idx);
  k_gemm<2,2><<<dim3(M32, 16), 32, 0, stream>>>(Hh + 128, 512, We4, 128, 128,
                                                T, (f16*)nullptr, 512,
                                                nullptr, nullptr, 0);
  k_agg<<<(MTOT * 64 + 255) / 256, 256, 0, stream>>>(T, 256, idx, SC + 768, BI + 768,
                                                     Hf, Hh, 256);

  // ---- point MLPs on concat(512) with fused BN+LReLU epilogue ----
  k_gemm<2,2><<<dim3(M32, 8), 32, 0, stream>>>(Hh, 512, Wp5, 512, 512,
                                               (float*)nullptr, P5h, 256,
                                               SC + 4 * 256, BI + 4 * 256, 1);
  k_gemm<2,2><<<dim3(M32, 4), 32, 0, stream>>>(P5h, 256, Wp6, 256, 256,
                                               (float*)nullptr, P6h, 128,
                                               SC + 5 * 256, BI + 5 * 256, 1);
  k_gemm<2,2><<<dim3(M32, 2), 32, 0, stream>>>(P6h, 128, Wp7, 128, 128,
                                               (float*)nullptr, P7h, 64,
                                               SC + 6 * 256, BI + 6 * 256, 1);
  // head: O=2 padded to 16
  k_gemm<2,1><<<dim3(M32, 1), 32, 0, stream>>>(P7h, 64, Wp8, 64, 64,
                                               Y8, (f16*)nullptr, 16,
                                               nullptr, nullptr, 0);
  k_writeout<<<(BATCH * 2 * NPTS + 255) / 256, 256, 0, stream>>>(Y8, out);
}
